// LocalWindowSelfAttention_2723009266021
// MI455X (gfx1250) — matrix operand order repaired and checked by reading, not compiled, not run on hardware
//
#include <hip/hip_runtime.h>

typedef __attribute__((ext_vector_type(16))) __bf16 v16bf;
typedef __attribute__((ext_vector_type(8)))  __bf16 v8bf;
typedef __attribute__((ext_vector_type(8)))  float  v8f;

namespace {
constexpr int kH = 32, kW = 32, kC = 256;
constexpr int kPad = 3, kWP = 38, kP = 1444;        // 38*38 padded pixels
constexpr int kNH = 8, kHD = 32, kNQ = 1024, kB = 4;
constexpr int kKV = 2 * kC;                          // 512
constexpr float kScale = 0.17677669529663687f;       // 32^-0.5
constexpr int kWinCols = 22;                         // 16 queries + 6 halo
constexpr int kWinRows = 7;
constexpr int kWinReal = kWinRows * kWinCols;        // 154
constexpr int kWin = 160;                            // padded to 32 multiple
constexpr int kM1 = kB * kP;                         // 5776 = 361*16
constexpr int kM2 = kB * kNQ;                        // 4096
}

// Async Global->LDS copy, 16B per lane. Per CDNA5 ISA the
// instruction offset is added to BOTH the LDS destination and global source
// address, so one base pair + offsets covers a contiguous row. Tracked by
// ASYNCcnt; fenced with s_wait_asynccnt.
#define ASYNC_B128(ldsa, ga, OFF)                                          \
  asm volatile("global_load_async_to_lds_b128 %0, %1, off offset:" #OFF    \
               :: "v"(ldsa), "v"(ga) : "memory")

__device__ inline void wait_asynccnt0() {
  asm volatile("s_wait_asynccnt 0" ::: "memory");
}

// Build a 16x32 bf16 A-fragment: lane (row = l&15, half = l>>4)
// holds K = half*16 + [0..15] (matching the B-fragments) -> two contiguous
// 8-halfword (16B) runs.
__device__ inline v16bf make_a_frag(const __bf16* lo, const __bf16* hi) {
  v8bf l = *(const v8bf*)lo;
  v8bf h = *(const v8bf*)hi;
  v16bf a;
#pragma unroll
  for (int e = 0; e < 8; ++e) { a[e] = l[e]; a[8 + e] = h[e]; }
  return a;
}

// ---------------------------------------------------------------------------
// Prep kernels
// ---------------------------------------------------------------------------
__global__ void pad_flatten_bf16(const float* __restrict__ batch, __bf16* __restrict__ X) {
  const int row = blockIdx.x;           // b*kP + p
  const int c = threadIdx.x;            // 256 channels
  const int b = row / kP;
  const int p = row - b * kP;
  const int pr = p / kWP;
  const int pc = p - pr * kWP;
  float v = 0.0f;
  if (pr >= kPad && pr < kPad + kH && pc >= kPad && pc < kPad + kW)
    v = batch[(((size_t)b * kC + c) * kH + (pr - kPad)) * kW + (pc - kPad)];
  X[(size_t)row * kC + c] = (__bf16)v;
}

__global__ void cvt_bf16(const float* __restrict__ src, __bf16* __restrict__ dst, int n) {
  int idx = blockIdx.x * blockDim.x + threadIdx.x;
  if (idx < n) dst[idx] = (__bf16)src[idx];
}

// dst[n*K + k] = src[k*N + n]   (weights stored transposed so B-fragments are
// contiguous 32B loads)
__global__ void transpose_bf16(const float* __restrict__ src, __bf16* __restrict__ dst,
                               int K, int N) {
  int idx = blockIdx.x * blockDim.x + threadIdx.x;
  if (idx >= K * N) return;
  int k = idx / N;
  int n = idx - k * N;
  dst[(size_t)n * K + k] = (__bf16)src[idx];
}

// ---------------------------------------------------------------------------
// Generic K=256 WMMA GEMM: one 16x16 tile per wave, 4 waves per block.
// A: bf16 [M x 256] row-major.  Bt: bf16 [N x 256] (i.e. B transposed).
// mode 0: Obf[m*N + n] = bf16((acc+bias)*scale)
// mode 1: Of32[(b*kC + n)*kNQ + q] = (acc+bias)*scale   (final [B,C,H,W] write)
// ---------------------------------------------------------------------------
__global__ __launch_bounds__(128)
void gemm16_wmma(const __bf16* __restrict__ A, const __bf16* __restrict__ Bt,
                 const float* __restrict__ bias, float scale, int N,
                 __bf16* __restrict__ Obf, float* __restrict__ Of32, int mode) {
  const int lane = threadIdx.x & 31;
  const int wv = threadIdx.x >> 5;
  const int ln = lane & 15;
  const int half = lane >> 4;
  const int n0 = (blockIdx.x * 4 + wv) * 16;
  const int m0 = blockIdx.y * 16;

  const __bf16* arow = A + (size_t)(m0 + ln) * kC;
  const __bf16* brow = Bt + (size_t)(n0 + ln) * kC;
  const int abase = half * 16;
  const int bbase = half * 16;

  v8f acc = {};
#pragma unroll
  for (int k0 = 0; k0 < kC; k0 += 32) {
    v16bf a = make_a_frag(arow + k0 + abase, arow + k0 + abase + 8);
    // B 32x16 fragment: lane (col = l&15, half) holds K = half*16 + [0..15],
    // contiguous in Bt row.
    v16bf b = *(const v16bf*)(brow + k0 + bbase);
    acc = __builtin_amdgcn_wmma_f32_16x16x32_bf16(false, a, false, b, (short)0,
                                                  acc, false, false);
  }

  const float bv = bias[n0 + ln];
#pragma unroll
  for (int j = 0; j < 8; ++j) {
    const int m = m0 + j + 8 * half;                // D row for VGPR j
    const float val = (acc[j] + bv) * scale;
    if (mode == 0) {
      Obf[(size_t)m * N + n0 + ln] = (__bf16)val;
    } else {
      const int bb = m >> 10;                       // m / kNQ
      const int q = m & 1023;
      Of32[((size_t)bb * kC + (n0 + ln)) * kNQ + q] = val;
    }
  }
}

// ---------------------------------------------------------------------------
// Windowed attention: one wave per (b, head, 16-query tile).
// 16 queries share one image row -> KV window = 7 padded rows x 22 padded cols
// = 154 positions (padded to 160 = 10 WMMA N-tiles / 5 K=32 chunks).
// K window staged via the async Global->LDS engine (ASYNCcnt); V window staged
// transposed via VGPRs (the AV B-fragment needs V columns contiguous in LDS),
// overlapping with the in-flight K DMA.
// ---------------------------------------------------------------------------
__device__ inline bool window_valid(int w, int i, int j0, int r) {
  if (w >= kWinReal) return false;
  const int rw = w / kWinCols;
  const int cw = w - rw * kWinCols;
  const int pr = i + rw;                 // padded row
  const int pc = j0 + cw;                // padded col
  return (pr >= kPad) && (pr < kPad + kH) &&
         (cw >= r) && (cw < r + 7) &&
         (pc >= kPad) && (pc < kPad + kW);
}

__global__ __launch_bounds__(32)
void attn_wmma(const __bf16* __restrict__ KV, const __bf16* __restrict__ Qh,
               __bf16* __restrict__ AttO) {
  __shared__ alignas(32) __bf16 kwin[kWin][kHD];    // K window, row-major
  __shared__ alignas(32) __bf16 vwinT[kHD][kWin];   // V window, transposed
  __shared__ alignas(32) __bf16 at[16][kWin];       // attention probs (bf16)
  __shared__ alignas(16) float  sc[16][kWin];       // scores / exp scratch

  const int lane = threadIdx.x;
  const int ln = lane & 15;
  const int half = lane >> 4;
  const int b = blockIdx.z;
  const int h = blockIdx.y;
  const int q0 = blockIdx.x * 16;
  const int i = q0 >> 5;                 // query image row
  const int j0 = q0 & 31;                // 0 or 16

  // ---- stage K (async DMA to LDS) and V (transposed, via VGPRs) ----
  for (int r5 = 0; r5 < 5; ++r5) {
    const int w = r5 * 32 + lane;
    if (w < kWinReal) {
      const int rw = w / kWinCols;
      const int cw = w - rw * kWinCols;
      const int p = (i + rw) * kWP + (j0 + cw);
      const __bf16* src = KV + ((size_t)(b * kP + p)) * kKV + h * kHD;
      // K row: 64B straight to LDS, no VGPR round-trip.
      const unsigned int lk = (unsigned int)(uintptr_t)&kwin[w][0];
      const unsigned long long gk = (unsigned long long)(uintptr_t)src;
      ASYNC_B128(lk, gk, 0);
      ASYNC_B128(lk, gk, 16);
      ASYNC_B128(lk, gk, 32);
      ASYNC_B128(lk, gk, 48);
      // V row: load and scatter transposed (overlaps with async K DMA).
      v16bf v0 = *(const v16bf*)(src + kC);
      v16bf v1 = *(const v16bf*)(src + kC + 16);
#pragma unroll
      for (int d = 0; d < 16; ++d) { vwinT[d][w] = v0[d]; vwinT[16 + d][w] = v1[d]; }
    } else {
      v16bf z = {};
      *(v16bf*)&kwin[w][0] = z;
      *(v16bf*)&kwin[w][16] = z;
#pragma unroll
      for (int d = 0; d < 32; ++d) vwinT[d][w] = (__bf16)0.0f;
    }
  }
  wait_asynccnt0();                      // K DMA complete
  __syncthreads();

  // ---- scores: Q[16x32] @ K^T[32x160] -> sc ----
  const __bf16* qrow = Qh + ((size_t)(b * kNQ + q0 + ln)) * kC + h * kHD;
  const v16bf qa = make_a_frag(qrow + half * 16, qrow + half * 16 + 8);
#pragma unroll
  for (int t = 0; t < kWin / 16; ++t) {
    v16bf kb = *(const v16bf*)&kwin[16 * t + ln][half * 16];
    v8f c = {};
    c = __builtin_amdgcn_wmma_f32_16x16x32_bf16(false, qa, false, kb, (short)0,
                                                c, false, false);
#pragma unroll
    for (int j = 0; j < 8; ++j) sc[j + 8 * half][16 * t + ln] = c[j];
  }
  __syncthreads();

  // ---- masked softmax: 2 lanes per query row (80 cols each) ----
  {
    const int r = ln;                    // query-local row
    const int c0 = half * (kWin / 2);
    float mx = -3.0e38f;
    for (int wi = 0; wi < kWin / 2; ++wi) {
      const int w = c0 + wi;
      if (window_valid(w, i, j0, r)) mx = fmaxf(mx, sc[r][w]);
    }
    mx = fmaxf(mx, __shfl_xor(mx, 16, 32));
    float sum = 0.0f;
    for (int wi = 0; wi < kWin / 2; ++wi) {
      const int w = c0 + wi;
      const float e = window_valid(w, i, j0, r) ? __expf(sc[r][w] - mx) : 0.0f;
      sc[r][w] = e;
      sum += e;
    }
    sum += __shfl_xor(sum, 16, 32);
    const float rinv = 1.0f / sum;
    for (int wi = 0; wi < kWin / 2; ++wi) {
      const int w = c0 + wi;
      at[r][w] = (__bf16)(sc[r][w] * rinv);
    }
  }
  __syncthreads();

  // ---- out: attn[16x160] @ V[160x32] ----
#pragma unroll
  for (int nh2 = 0; nh2 < 2; ++nh2) {
    v8f acc = {};
#pragma unroll
    for (int kc = 0; kc < 5; ++kc) {
      v16bf a = make_a_frag(&at[ln][32 * kc + half * 16],
                            &at[ln][32 * kc + half * 16 + 8]);
      v16bf bv = *(const v16bf*)&vwinT[nh2 * 16 + ln][32 * kc + half * 16];
      acc = __builtin_amdgcn_wmma_f32_16x16x32_bf16(false, a, false, bv, (short)0,
                                                    acc, false, false);
    }
#pragma unroll
    for (int j = 0; j < 8; ++j) {
      const int m = j + 8 * half;
      AttO[((size_t)(b * kNQ + q0 + m)) * kC + h * kHD + nh2 * 16 + ln] =
          (__bf16)acc[j];
    }
  }
}

// ---------------------------------------------------------------------------
extern "C" void kernel_launch(void* const* d_in, const int* in_sizes, int n_in,
                              void* d_out, int out_size, void* d_ws, size_t ws_size,
                              hipStream_t stream) {
  (void)in_sizes; (void)n_in; (void)out_size; (void)ws_size;
  const float* batch = (const float*)d_in[0];
  const float* query = (const float*)d_in[1];
  const float* W_kv  = (const float*)d_in[2];
  const float* b_kv  = (const float*)d_in[3];
  const float* W_q   = (const float*)d_in[4];
  const float* b_q   = (const float*)d_in[5];
  const float* W_out = (const float*)d_in[6];
  const float* b_out = (const float*)d_in[7];
  float* out = (float*)d_out;

  char* ws = (char*)d_ws;
  size_t off = 0;
  auto alloc = [&](size_t bytes) -> char* {
    char* p = ws + off;
    off += (bytes + 255) & ~(size_t)255;
    return p;
  };
  __bf16* X     = (__bf16*)alloc((size_t)kM1 * kC * 2);   // padded flat input
  __bf16* Wkvt  = (__bf16*)alloc((size_t)kKV * kC * 2);   // W_kv^T
  __bf16* Wqt   = (__bf16*)alloc((size_t)kC * kC * 2);    // W_q^T
  __bf16* Woutt = (__bf16*)alloc((size_t)kC * kC * 2);    // W_out^T
  __bf16* Qx    = (__bf16*)alloc((size_t)kM2 * kC * 2);   // query bf16
  __bf16* KVb   = (__bf16*)alloc((size_t)kM1 * kKV * 2);  // k|v per pixel
  __bf16* Qh    = (__bf16*)alloc((size_t)kM2 * kC * 2);   // projected q
  __bf16* AttO  = (__bf16*)alloc((size_t)kM2 * kC * 2);   // attention output

  pad_flatten_bf16<<<kM1, kC, 0, stream>>>(batch, X);
  cvt_bf16<<<(kM2 * kC + 255) / 256, 256, 0, stream>>>(query, Qx, kM2 * kC);
  transpose_bf16<<<(kC * kKV + 255) / 256, 256, 0, stream>>>(W_kv, Wkvt, kC, kKV);
  transpose_bf16<<<(kC * kC + 255) / 256, 256, 0, stream>>>(W_q, Wqt, kC, kC);
  transpose_bf16<<<(kC * kC + 255) / 256, 256, 0, stream>>>(W_out, Woutt, kC, kC);

  // KV projection: [5776 x 256] @ [256 x 512]
  gemm16_wmma<<<dim3(kKV / 64, kM1 / 16), 128, 0, stream>>>(
      X, Wkvt, b_kv, 1.0f, kKV, KVb, nullptr, 0);
  // Q projection (fused *SCALE): [4096 x 256] @ [256 x 256]
  gemm16_wmma<<<dim3(kC / 64, kM2 / 16), 128, 0, stream>>>(
      Qx, Wqt, b_q, kScale, kC, Qh, nullptr, 0);
  // Windowed attention
  attn_wmma<<<dim3(kNQ / 16, kNH, kB), 32, 0, stream>>>(KVb, Qh, AttO);
  // Output projection, writes final [B,C,H,W] f32
  gemm16_wmma<<<dim3(kC / 64, kM2 / 16), 128, 0, stream>>>(
      AttO, Woutt, b_out, 1.0f, kC, nullptr, out, 1);
}
